// RNNCRF_Unary_Labeler_32452772888625
// MI455X (gfx1250) — compile-verified, matrix operand order checked by reading, not run on hardware
//
#include <hip/hip_runtime.h>
#include <hip/hip_bf16.h>

// ---------------------------------------------------------------------------
// RNN-CRF labeler for MI455X (gfx1250, wave32).
//   k_cvt_bf16      : fp32 -> bf16 (RNE) for seq / W_ih / W_hh / W_tag
//   k_prep          : bias = b_ih + b_hh; reset grid-sync counter
//   k_xgates_wmma   : x_gates = seq @ W_ih^T + bias     (v_wmma_f32_16x16x32_bf16)
//   k_lstm          : persistent 32-WGP recurrence; W_hh slice DMA'd into
//                     320KB LDS via TENSOR_LOAD_TO_LDS (TDM), resident for all
//                     8192 steps; device-scope rel/acq counter = grid barrier
//   k_feat_wmma     : feat = hs @ W_tag^T + b_tag       (v_wmma_f32_16x16x32_bf16)
//   k_crf           : sequential log-semiring alpha recursion (K=16), 1 wave
// ---------------------------------------------------------------------------

#define TT   8192
#define DD   1024
#define HH   1024
#define G4   4096
#define KK   16
#define NWG  32
#define NEGINF (-100000.0f)

typedef __attribute__((ext_vector_type(16))) __bf16          v16bf;
typedef __attribute__((ext_vector_type(8)))  float           v8f;
typedef __attribute__((ext_vector_type(8)))  unsigned short  v8us;
typedef __attribute__((ext_vector_type(16))) unsigned short  v16us;
typedef __attribute__((ext_vector_type(4)))  unsigned int    v4u;
typedef __attribute__((ext_vector_type(8)))  int             v8i;
typedef __attribute__((ext_vector_type(4)))  int             v4i;

__device__ __forceinline__ unsigned short f32_to_bf16_rne(float f) {
  unsigned u = __float_as_uint(f);
  u += 0x7fffu + ((u >> 16) & 1u);
  return (unsigned short)(u >> 16);
}
__device__ __forceinline__ float bf16_to_f32(unsigned short s) {
  return __uint_as_float(((unsigned)s) << 16);
}

// ---- TDM: DMA `nrows` contiguous rows of `row_bytes` from global into LDS.
// Builds a 2-D Tensor DMA Descriptor (D#) per CDNA5 ISA:
//   group0: count=1 | lds_addr | global_addr | type=2
//   group1: data_size=3 (8B units); tensor_dim0=tile_dim0=row units;
//           tensor_dim1=tile_dim1=nrows; stride0=row units; no pad/multicast
//   groups 2/3: zero (<=2D tensor)
__device__ __forceinline__ void tdm_load_rows_to_lds(const void* gptr,
                                                     unsigned lds_off_bytes,
                                                     unsigned row_bytes,
                                                     unsigned nrows) {
  const unsigned long long ga = (unsigned long long)(uintptr_t)gptr;
  const unsigned units = row_bytes >> 3;  // 8-byte units per row
  v4u g0;
  g0[0] = 1u;                                    // count=1 (valid), load
  g0[1] = lds_off_bytes;                         // lds_addr (bytes)
  g0[2] = (unsigned)(ga & 0xffffffffu);          // global_addr[31:0]
  g0[3] = (unsigned)((ga >> 32) & 0x1ffffffu)    // global_addr[56:32]
        | (2u << 30);                            // type = 2 ("image")
  v8i g1;
  g1[0] = (int)(3u << 16);                       // wg_mask=0, data_size=3 (8B)
  g1[1] = (int)(units << 16);                    // tensor_dim0[15:0] @ [63:48]
  g1[2] = (int)(nrows << 16);                    // tensor_dim1[15:0] @ [95:80]
  g1[3] = (int)(units << 16);                    // tile_dim0        @ [127:112]
  g1[4] = (int)nrows;                            // tile_dim1        @ [143:128]
  g1[5] = (int)units;                            // tensor_dim0_stride[31:0]
  g1[6] = 0;
  g1[7] = 0;
  v4i z4 = {0, 0, 0, 0};
#if defined(__clang_major__) && (__clang_major__ >= 23)
  v8i z8 = {0, 0, 0, 0, 0, 0, 0, 0};
  __builtin_amdgcn_tensor_load_to_lds(g0, g1, z4, z4, z8, 0);
#else
  __builtin_amdgcn_tensor_load_to_lds(g0, g1, z4, z4, 0);
#endif
}

// Load one 16x32 bf16 WMMA fragment (A layout; B mirrors it: lane holds the
// output column's row of W, contiguous along K). Two 16B contiguous loads.
__device__ __forceinline__ v16bf load_frag16x32(const unsigned short* base,
                                                int ld, int row, int k0, int lane) {
  const int m  = lane & 15;
  const int kb = k0 + ((lane >> 4) << 3);
  const unsigned short* p = base + (size_t)(row + m) * ld + kb;
  v8us lo = *(const v8us*)(p);
  v8us hi = *(const v8us*)(p + 16);
  v16us u;
#pragma unroll
  for (int i = 0; i < 8; ++i) { u[i] = lo[i]; u[8 + i] = hi[i]; }
  return __builtin_bit_cast(v16bf, u);
}

// ---------------------------------------------------------------- converters
__global__ void k_cvt_bf16(const float* __restrict__ src,
                           unsigned short* __restrict__ dst, int n) {
  int i = blockIdx.x * blockDim.x + threadIdx.x;
  if (i < n) dst[i] = f32_to_bf16_rne(src[i]);
}

__global__ void k_prep(const float* __restrict__ b_ih, const float* __restrict__ b_hh,
                       float* __restrict__ bias, unsigned int* __restrict__ ctr) {
  int i = blockIdx.x * blockDim.x + threadIdx.x;
  if (i < G4) bias[i] = b_ih[i] + b_hh[i];
  if (i == 0) *ctr = 0u;
}

// ------------------------------------------------- x_gates = seq @ W_ih^T + b
// block = 256 threads = 8 waves; wave -> 16 rows x 64 cols (4 accumulators)
__global__ void k_xgates_wmma(const unsigned short* __restrict__ Xb,
                              const unsigned short* __restrict__ Wb,
                              const float* __restrict__ bias,
                              float* __restrict__ xg) {
  const int lane  = threadIdx.x & 31;
  const int wave  = threadIdx.x >> 5;
  const int mbase = blockIdx.x * 128 + wave * 16;
  const int nbase = blockIdx.y * 64;
  v8f acc[4];
#pragma unroll
  for (int c = 0; c < 4; ++c)
#pragma unroll
    for (int v = 0; v < 8; ++v) acc[c][v] = 0.0f;

  for (int k0 = 0; k0 < DD; k0 += 32) {
    v16bf a = load_frag16x32(Xb, DD, mbase, k0, lane);
#pragma unroll
    for (int c = 0; c < 4; ++c) {
      v16bf b = load_frag16x32(Wb, DD, nbase + c * 16, k0, lane);
      acc[c] = __builtin_amdgcn_wmma_f32_16x16x32_bf16(
          false, a, false, b, (short)0, acc[c], false, false);
    }
  }
#pragma unroll
  for (int c = 0; c < 4; ++c) {
    const int n  = nbase + c * 16 + (lane & 15);
    const float bv = bias[n];
#pragma unroll
    for (int v = 0; v < 8; ++v) {
      const int m = mbase + v + ((lane >> 4) << 3);
      xg[(size_t)m * G4 + n] = acc[c][v] + bv;
    }
  }
}

// ------------------------------------------------------- persistent LSTM scan
// 32 blocks x 1024 threads. Block b owns h indices [b*32, b*32+32) and the 128
// matching W_hh rows (i,f,g,o), DMA'd by the TDM into LDS (256KB bf16) and
// resident for ALL timesteps. Per step: 8 threads/row do partial dots over
// LDS, shuffle-reduce, gate math, publish 32 h values; device-scope counter
// gives the grid barrier.
__global__ void __launch_bounds__(1024, 1)
k_lstm(const unsigned short* __restrict__ Whh, const float* __restrict__ xg,
       float* __restrict__ hbuf, unsigned short* __restrict__ hs,
       unsigned int* __restrict__ ctr) {
  extern __shared__ unsigned char smem[];
  unsigned short* w  = (unsigned short*)smem;                     // 128*1024 bf16
  float*          hsh = (float*)(smem + 128 * HH * 2);            // 1024 f32
  float*          gsh = (float*)(smem + 128 * HH * 2 + HH * 4);   // 128 f32
  float*          csh = gsh + 128;                                // 32 f32

  const int tid   = threadIdx.x;
  const int jbase = blockIdx.x * 32;
  const int wv    = tid >> 5;  // wave id (uniform per wave)

  // TDM preload: wave g DMAs gate-chunk g (32 rows x 2048B) into LDS.
  if (wv < 4) {
    tdm_load_rows_to_lds(Whh + (size_t)(wv * HH + jbase) * HH,
                         __builtin_amdgcn_groupstaticsize() +
                             (unsigned)(wv * 32 * HH * 2),
                         HH * 2, 32);
  }
  __builtin_amdgcn_s_wait_tensorcnt(0);  // no-op for non-issuing waves
  if (tid < 32) csh[tid] = 0.0f;
  __syncthreads();

  const int rl   = tid >> 3;   // 0..127 local gate-row
  const int part = tid & 7;    // 0..7 K-partition
  const int g    = rl >> 5, jj = rl & 31;
  const int grow = g * HH + jbase + jj;  // column in x_gates
  const unsigned short* wrow = w + rl * HH;

  for (int t = 0; t < TT; ++t) {
    // ---- acquire h_{t-1}
    if (t == 0) {
      hsh[tid] = 0.0f;
    } else {
      if (tid == 0) {
        while (__hip_atomic_load(ctr, __ATOMIC_ACQUIRE,
                                 __HIP_MEMORY_SCOPE_AGENT) < (unsigned)(t * NWG))
          __builtin_amdgcn_s_sleep(2);
      }
      __syncthreads();
      hsh[tid] = __hip_atomic_load(&hbuf[((t - 1) & 1) * HH + tid],
                                   __ATOMIC_RELAXED, __HIP_MEMORY_SCOPE_AGENT);
    }
    if (t + 1 < TT) __builtin_prefetch(&xg[(size_t)(t + 1) * G4 + grow], 0, 1);
    __syncthreads();

    // ---- partial dot over LDS-resident W_hh (16B bf16 chunks, interleaved K)
    float acc = 0.0f;
#pragma unroll 4
    for (int kk = 0; kk < 16; ++kk) {
      const int k = kk * 64 + part * 8;
      v8us wvv = *(const v8us*)(wrow + k);
#pragma unroll
      for (int e = 0; e < 8; ++e)
        acc = fmaf(bf16_to_f32(wvv[e]), hsh[k + e], acc);
    }
    acc += __shfl_down(acc, 4, 8);
    acc += __shfl_down(acc, 2, 8);
    acc += __shfl_down(acc, 1, 8);
    if (part == 0) gsh[rl] = acc + xg[(size_t)t * G4 + grow];
    __syncthreads();

    // ---- gate math + publish h_t
    if (tid < 32) {
      const float gi = gsh[0 * 32 + tid], gf = gsh[1 * 32 + tid];
      const float gg = gsh[2 * 32 + tid], go = gsh[3 * 32 + tid];
      const float si = 1.0f / (1.0f + expf(-gi));
      const float sf = 1.0f / (1.0f + expf(-gf));
      const float so = 1.0f / (1.0f + expf(-go));
      const float c  = sf * csh[tid] + si * tanhf(gg);
      csh[tid] = c;
      const float h = so * tanhf(c);
      const int j = jbase + tid;
      __hip_atomic_store(&hbuf[(t & 1) * HH + j], h, __ATOMIC_RELAXED,
                         __HIP_MEMORY_SCOPE_AGENT);
      hs[(size_t)t * HH + j] = f32_to_bf16_rne(h);
    }
    __threadfence();
    __syncthreads();
    if (tid == 0)
      __hip_atomic_fetch_add(ctr, 1u, __ATOMIC_RELEASE, __HIP_MEMORY_SCOPE_AGENT);
  }
}

// -------------------------------------------------- feat = hs @ W_tag^T + b
// block = 256 threads = 8 waves; wave -> 16 rows x 16 cols (single N tile)
__global__ void k_feat_wmma(const unsigned short* __restrict__ Hb,
                            const unsigned short* __restrict__ Wt,
                            const float* __restrict__ b_tag,
                            float* __restrict__ feat) {
  const int lane  = threadIdx.x & 31;
  const int wave  = threadIdx.x >> 5;
  const int mbase = blockIdx.x * 128 + wave * 16;
  v8f acc;
#pragma unroll
  for (int v = 0; v < 8; ++v) acc[v] = 0.0f;

  for (int k0 = 0; k0 < HH; k0 += 32) {
    v16bf a = load_frag16x32(Hb, HH, mbase, k0, lane);
    v16bf b = load_frag16x32(Wt, HH, 0, k0, lane);
    acc = __builtin_amdgcn_wmma_f32_16x16x32_bf16(
        false, a, false, b, (short)0, acc, false, false);
  }
  const int n  = lane & 15;
  const float bv = b_tag[n];
#pragma unroll
  for (int v = 0; v < 8; ++v) {
    const int m = mbase + v + ((lane >> 4) << 3);
    feat[(size_t)m * KK + n] = acc[v] + bv;
  }
}

// ----------------------------------------------------------- CRF alpha scan
__global__ void k_crf(const float* __restrict__ feat,
                      const float* __restrict__ trans,
                      float* __restrict__ out) {
  __shared__ float tr[KK * KK];
  __shared__ float a[KK];
  __shared__ float anew[KK];
  const int tid = threadIdx.x;
  for (int i = tid; i < KK * KK; i += 32) tr[i] = trans[i];
  if (tid < KK) {
    a[tid] = (tid == 0) ? 0.0f : NEGINF;
    out[1 + tid] = a[tid];  // alpha0 row
  }
  __syncthreads();
  for (int t = 0; t < TT; ++t) {
    if (tid < KK) {
      float m = -3.4e38f;
#pragma unroll
      for (int i = 0; i < KK; ++i) m = fmaxf(m, a[i] + tr[i * KK + tid]);
      float s = 0.0f;
#pragma unroll
      for (int i = 0; i < KK; ++i) s += expf(a[i] + tr[i * KK + tid] - m);
      const float an = feat[(size_t)t * KK + tid] + m + logf(s);
      anew[tid] = an;
      out[1 + (size_t)(t + 1) * KK + tid] = an;
    }
    __syncthreads();
    if (tid < KK) a[tid] = anew[tid];
    __syncthreads();
  }
  if (tid == 0) {
    float m = -3.4e38f;
    for (int i = 0; i < KK; ++i) m = fmaxf(m, a[i]);
    float s = 0.0f;
    for (int i = 0; i < KK; ++i) s += expf(a[i] - m);
    out[0] = m + logf(s);  // Z
  }
}

// ---------------------------------------------------------------------------
extern "C" void kernel_launch(void* const* d_in, const int* in_sizes, int n_in,
                              void* d_out, int out_size, void* d_ws, size_t ws_size,
                              hipStream_t stream) {
  (void)in_sizes; (void)n_in; (void)out_size; (void)ws_size;
  const float* seq   = (const float*)d_in[0];  // (1, T, D)
  const float* W_ih  = (const float*)d_in[1];  // (4H, D)
  const float* W_hh  = (const float*)d_in[2];  // (4H, H)
  const float* b_ih  = (const float*)d_in[3];  // (4H,)
  const float* b_hh  = (const float*)d_in[4];  // (4H,)
  const float* W_tag = (const float*)d_in[5];  // (K, H)
  const float* b_tag = (const float*)d_in[6];  // (K,)
  const float* trans = (const float*)d_in[7];  // (K, K)
  float* out = (float*)d_out;                  // [Z | alpha (T+1,K)]

  // workspace layout (256B aligned)
  unsigned char* ws = (unsigned char*)d_ws;
  size_t off = 0;
  auto take = [&](size_t bytes) -> unsigned char* {
    unsigned char* p = ws + off;
    off = (off + bytes + 255) & ~(size_t)255;
    return p;
  };
  unsigned short* seq_b  = (unsigned short*)take((size_t)TT * DD * 2);
  unsigned short* wih_b  = (unsigned short*)take((size_t)G4 * DD * 2);
  unsigned short* whh_b  = (unsigned short*)take((size_t)G4 * HH * 2);
  unsigned short* wtag_b = (unsigned short*)take((size_t)KK * HH * 2);
  float*          bias   = (float*)take((size_t)G4 * 4);
  float*          xg     = (float*)take((size_t)TT * G4 * 4);
  unsigned short* hs     = (unsigned short*)take((size_t)TT * HH * 2);
  float*          hbuf   = (float*)take((size_t)2 * HH * 4);
  float*          feat   = (float*)take((size_t)TT * KK * 4);
  unsigned int*   ctr    = (unsigned int*)take(256);

  const int cvtB = 256;
  k_cvt_bf16<<<((TT * DD) + cvtB - 1) / cvtB, cvtB, 0, stream>>>(seq,   seq_b,  TT * DD);
  k_cvt_bf16<<<((G4 * DD) + cvtB - 1) / cvtB, cvtB, 0, stream>>>(W_ih,  wih_b,  G4 * DD);
  k_cvt_bf16<<<((G4 * HH) + cvtB - 1) / cvtB, cvtB, 0, stream>>>(W_hh,  whh_b,  G4 * HH);
  k_cvt_bf16<<<((KK * HH) + cvtB - 1) / cvtB, cvtB, 0, stream>>>(W_tag, wtag_b, KK * HH);
  k_prep<<<(G4 + cvtB - 1) / cvtB, cvtB, 0, stream>>>(b_ih, b_hh, bias, ctr);

  dim3 gx(TT / 128, G4 / 64);
  k_xgates_wmma<<<gx, 256, 0, stream>>>(seq_b, wih_b, bias, xg);

  const size_t lstm_lds = (size_t)128 * HH * 2 + HH * 4 + 128 * 4 + 32 * 4;
  k_lstm<<<NWG, 1024, lstm_lds, stream>>>(whh_b, xg, hbuf, hs, ctr);

  k_feat_wmma<<<TT / 128, 256, 0, stream>>>(hs, wtag_b, b_tag, feat);

  k_crf<<<1, 32, 0, stream>>>(feat, trans, out);
}
